// CanonicalMLP_49993419325744
// MI455X (gfx1250) — compile-verified, matrix operand order checked by reading, not run on hardware
//
#include <hip/hip_runtime.h>
#include <hip/hip_bf16.h>

typedef __bf16 bf16_t;
typedef __attribute__((ext_vector_type(16))) __bf16 v16bf;
typedef __attribute__((ext_vector_type(8)))  float  v8f;
typedef int v4i_vs __attribute__((vector_size(16)));   // matches builtin param type

#define K_NN   20
#define LRELU  0.2f
#define B_SZ   32
#define N_PTS  1024
#define BN_TOT (B_SZ * N_PTS)

// ---------------- CDNA5 async LDS copy helpers ----------------
__device__ inline void async_copy16(const void* g, void* l) {
#if __has_builtin(__builtin_amdgcn_global_load_async_to_lds_b128)
  __builtin_amdgcn_global_load_async_to_lds_b128(
      (__attribute__((address_space(1))) v4i_vs*)(void*)g,
      (__attribute__((address_space(3))) v4i_vs*)l, 0, 0);
#else
  *(uint4*)l = *(const uint4*)g;   // synchronous fallback (ds_store_b128)
#endif
}

__device__ inline void async_wait_le(int n) {
#if __has_builtin(__builtin_amdgcn_global_load_async_to_lds_b128)
#if __has_builtin(__builtin_amdgcn_s_wait_asynccnt)
  if (n == 0) __builtin_amdgcn_s_wait_asynccnt(0);
  else        __builtin_amdgcn_s_wait_asynccnt(1);
#else
  if (n == 0) asm volatile("s_wait_asynccnt 0x0" ::: "memory");
  else        asm volatile("s_wait_asynccnt 0x1" ::: "memory");
#endif
#endif
}

// ---------------- wave helpers ----------------
__device__ inline float wsum32(float v) {
  for (int m = 16; m; m >>= 1) v += __shfl_xor(v, m, 32);
  return v;
}

__device__ inline float blockSum256(float v) {
  __shared__ float sh[8];
  __shared__ float total;
  int lane = threadIdx.x & 31, wid = threadIdx.x >> 5;
  for (int m = 16; m; m >>= 1) v += __shfl_xor(v, m, 32);
  __syncthreads();
  if (lane == 0) sh[wid] = v;
  __syncthreads();
  if (threadIdx.x == 0) {
    float s = 0.f;
    for (int i = 0; i < 8; ++i) s += sh[i];
    total = s;
  }
  __syncthreads();
  return total;
}

// ---------------- closed-form symmetric 3x3 eigensolver ----------------
__device__ inline void eig3_sym(float a00, float a01, float a02,
                                float a11, float a12, float a22, float R[9]) {
  float q  = (a00 + a11 + a22) * (1.f / 3.f);
  float p1 = a01 * a01 + a02 * a02 + a12 * a12;
  float p2 = (a00 - q) * (a00 - q) + (a11 - q) * (a11 - q) + (a22 - q) * (a22 - q) + 2.f * p1;
  float p  = sqrtf(fmaxf(p2 * (1.f / 6.f), 1e-30f));
  float ip = 1.f / p;
  float b00 = (a00 - q) * ip, b01 = a01 * ip, b02 = a02 * ip;
  float b11 = (a11 - q) * ip, b12 = a12 * ip, b22 = (a22 - q) * ip;
  float detB = b00 * (b11 * b22 - b12 * b12)
             - b01 * (b01 * b22 - b12 * b02)
             + b02 * (b01 * b12 - b11 * b02);
  float r   = fminf(fmaxf(0.5f * detB, -1.f), 1.f);
  float phi = acosf(r) * (1.f / 3.f);
  float e0  = q + 2.f * p * cosf(phi);
  float e2  = q + 2.f * p * cosf(phi + 2.0943951023931953f);
  float e1  = 3.f * q - e0 - e2;
  float ev[3] = {e0, e1, e2};
  for (int j = 0; j < 3; ++j) {
    float l = ev[j];
    float r0x = a00 - l, r0y = a01,     r0z = a02;
    float r1x = a01,     r1y = a11 - l, r1z = a12;
    float r2x = a02,     r2y = a12,     r2z = a22 - l;
    float c0x = r0y * r1z - r0z * r1y, c0y = r0z * r1x - r0x * r1z, c0z = r0x * r1y - r0y * r1x;
    float c1x = r0y * r2z - r0z * r2y, c1y = r0z * r2x - r0x * r2z, c1z = r0x * r2y - r0y * r2x;
    float c2x = r1y * r2z - r1z * r2y, c2y = r1z * r2x - r1x * r2z, c2z = r1x * r2y - r1y * r2x;
    float n0 = c0x * c0x + c0y * c0y + c0z * c0z;
    float n1 = c1x * c1x + c1y * c1y + c1z * c1z;
    float n2 = c2x * c2x + c2y * c2y + c2z * c2z;
    float vx = c0x, vy = c0y, vz = c0z, nn = n0;
    if (n1 > nn) { vx = c1x; vy = c1y; vz = c1z; nn = n1; }
    if (n2 > nn) { vx = c2x; vy = c2y; vz = c2z; nn = n2; }
    if (nn < 1e-24f) { vx = (j == 0); vy = (j == 1); vz = (j == 2); nn = 1.f; }
    float inv = rsqrtf(nn);
    R[0 * 3 + j] = vx * inv; R[1 * 3 + j] = vy * inv; R[2 * 3 + j] = vz * inv;
  }
  for (int j = 0; j < 3; ++j) {
    float a0 = R[j], a1 = R[3 + j], a2 = R[6 + j];
    float mv = a0, ma = fabsf(a0);
    if (fabsf(a1) > ma) { mv = a1; ma = fabsf(a1); }
    if (fabsf(a2) > ma) { mv = a2; }
    float s = (mv >= 0.f) ? 1.f : -1.f;
    R[j] *= s; R[3 + j] *= s; R[6 + j] *= s;
  }
  float c0x = R[0], c0y = R[3], c0z = R[6];
  float c1x = R[1], c1y = R[4], c1z = R[7];
  float c2x = R[2], c2y = R[5], c2z = R[8];
  float det = c0x * (c1y * c2z - c1z * c2y)
            - c0y * (c1x * c2z - c1z * c2x)
            + c0z * (c1x * c2y - c1y * c2x);
  if (det < 0.f) { R[2] = -R[2]; R[5] = -R[5]; R[8] = -R[8]; }
}

// ---------------- 1) full-cloud PCA canonicalization ----------------
__global__ void cloud_pca_kernel(const float* __restrict__ x, float* __restrict__ canon) {
  const int N = N_PTS, PPT = N_PTS / 256;
  int b = blockIdx.x, t = threadIdx.x;
  const float* xb = x + (size_t)b * 3 * N;
  float px[PPT], py[PPT], pz[PPT];
  float sx = 0, sy = 0, sz = 0;
  for (int i = 0; i < PPT; ++i) {
    int n = t + i * 256;
    px[i] = xb[n]; py[i] = xb[N + n]; pz[i] = xb[2 * N + n];
    sx += px[i]; sy += py[i]; sz += pz[i];
  }
  float mx = blockSum256(sx) / N, my = blockSum256(sy) / N, mz = blockSum256(sz) / N;
  float cxx = 0, cxy = 0, cxz = 0, cyy = 0, cyz = 0, czz = 0;
  for (int i = 0; i < PPT; ++i) {
    px[i] -= mx; py[i] -= my; pz[i] -= mz;
    cxx += px[i] * px[i]; cxy += px[i] * py[i]; cxz += px[i] * pz[i];
    cyy += py[i] * py[i]; cyz += py[i] * pz[i]; czz += pz[i] * pz[i];
  }
  float inc = 1.f / (N - 1);
  float a00 = blockSum256(cxx) * inc, a01 = blockSum256(cxy) * inc, a02 = blockSum256(cxz) * inc;
  float a11 = blockSum256(cyy) * inc, a12 = blockSum256(cyz) * inc, a22 = blockSum256(czz) * inc;
  float R[9];
  eig3_sym(a00, a01, a02, a11, a12, a22, R);
  float qx[PPT], qy[PPT], qz[PPT];
  float k3x = 0, k3y = 0, k3z = 0;
  for (int i = 0; i < PPT; ++i) {
    qx[i] = px[i] * R[0] + py[i] * R[3] + pz[i] * R[6];
    qy[i] = px[i] * R[1] + py[i] * R[4] + pz[i] * R[7];
    qz[i] = px[i] * R[2] + py[i] * R[5] + pz[i] * R[8];
    k3x += qx[i] * qx[i] * qx[i]; k3y += qy[i] * qy[i] * qy[i]; k3z += qz[i] * qz[i] * qz[i];
  }
  float skx = blockSum256(k3x), sky = blockSum256(k3y), skz = blockSum256(k3z);
  float s0 = (skx >= 0.f) ? 1.f : -1.f;
  float s1 = (sky >= 0.f) ? 1.f : -1.f;
  float s2 = (skz >= 0.f) ? 1.f : -1.f;
  int neg = (s0 < 0.f) + (s1 < 0.f) + (s2 < 0.f);
  if (neg & 1) s2 = -s2;
  for (int i = 0; i < PPT; ++i) {
    int n = t + i * 256;
    size_t g = ((size_t)b * N + n) * 3;
    canon[g + 0] = qx[i] * s0; canon[g + 1] = qy[i] * s1; canon[g + 2] = qz[i] * s2;
  }
}

// ---------------- 2) point-major padded bf16 feature tensors + norms ----------------
__global__ void make_xt_pts_kernel(const float* __restrict__ pts, bf16_t* __restrict__ XT,
                                   float* __restrict__ norms) {
  int bn = blockIdx.x * blockDim.x + threadIdx.x;
  if (bn >= BN_TOT) return;
  float a = pts[(size_t)bn * 3], b = pts[(size_t)bn * 3 + 1], c = pts[(size_t)bn * 3 + 2];
  bf16_t* row = XT + (size_t)bn * 32;
  row[0] = (bf16_t)a; row[1] = (bf16_t)b; row[2] = (bf16_t)c;
  for (int i = 3; i < 32; ++i) row[i] = (bf16_t)0.f;
  norms[bn] = a * a + b * b + c * c;
}

__global__ void make_xt_kernel(const bf16_t* __restrict__ X, int C, int cp,
                               bf16_t* __restrict__ XT, float* __restrict__ norms) {
  int bn = blockIdx.x * blockDim.x + threadIdx.x;
  if (bn >= BN_TOT) return;
  float s = 0.f;
  bf16_t* row = XT + (size_t)bn * cp;
  for (int c = 0; c < cp; ++c) {
    float v = (c < C) ? (float)X[(size_t)c * BN_TOT + bn] : 0.f;
    row[c] = (bf16_t)v;
    s += v * v;
  }
  norms[bn] = s;
}

// ---------------- 3) kNN: WMMA Gram strip + iterative wave argmax top-20 ----------------
__global__ void knn_kernel(const bf16_t* __restrict__ XT, const float* __restrict__ norms,
                           int cp, int* __restrict__ idx_out) {
  __shared__ float ld[16][N_PTS];   // 64 KB of the 320 KB WGP LDS
  int lane = threadIdx.x & 31;
  int wave = threadIdx.x >> 5;      // 4 waves
  int b = blockIdx.y;
  int r0 = blockIdx.x * 16;
  size_t base = (size_t)b * N_PTS;
  int arow = r0 + (lane & 15);
  int koff = (lane >> 4) * 16;
  const bf16_t* aBase = XT + (base + arow) * cp + koff;
  for (int ct = wave; ct < N_PTS / 16; ct += 4) {
    int c0 = ct * 16;
    const bf16_t* bBase = XT + (base + c0 + (lane & 15)) * cp + koff;
    v8f acc = {};
    for (int kc = 0; kc < cp; kc += 32) {
      v16bf av = *(const v16bf*)(aBase + kc);
      v16bf bv = *(const v16bf*)(bBase + kc);
      acc = __builtin_amdgcn_wmma_f32_16x16x32_bf16(false, av, false, bv,
                                                    (short)0, acc, false, false);
    }
    int col = lane & 15;
    float nc = norms[base + c0 + col];
#pragma unroll
    for (int v = 0; v < 8; ++v) {
      int r = v + 8 * (lane >> 4);
      ld[r][c0 + col] = 2.f * acc[v] - norms[base + r0 + r] - nc;  // neg_dist
    }
  }
  __syncthreads();
  for (int r = wave; r < 16; r += 4) {
    for (int t = 0; t < K_NN; ++t) {
      float best = -1e30f; int bi = 0;
      for (int j = lane; j < N_PTS; j += 32) {
        float v = ld[r][j];
        if (v > best) { best = v; bi = j; }
      }
      for (int m = 16; m; m >>= 1) {
        float ov = __shfl_xor(best, m, 32);
        int   oi = __shfl_xor(bi, m, 32);
        if (ov > best || (ov == best && oi < bi)) { best = ov; bi = oi; }
      }
      if (lane == 0) idx_out[(base + r0 + r) * K_NN + t] = bi;
      ld[r][bi] = -1e30f;
    }
  }
}

// ---------------- 4) per-patch PCA canonicalization (one wave32 / patch) ----------------
__global__ void patch_kernel(const float* __restrict__ canonPts, const int* __restrict__ idx,
                             const bf16_t* __restrict__ feat, int C,
                             bf16_t* __restrict__ P, int cip) {
  int lane = threadIdx.x & 31;
  size_t patch = (size_t)blockIdx.x * 4 + (threadIdx.x >> 5);
  int b = (int)(patch / N_PTS);
  bool act = lane < K_NN;
  int nbr = act ? idx[patch * K_NN + lane] : 0;
  size_t g = (size_t)b * N_PTS + nbr;
  float px = act ? canonPts[g * 3 + 0] : 0.f;
  float py = act ? canonPts[g * 3 + 1] : 0.f;
  float pz = act ? canonPts[g * 3 + 2] : 0.f;
  const float ik = 1.f / K_NN, ik1 = 1.f / (K_NN - 1);
  float mx = wsum32(px) * ik, my = wsum32(py) * ik, mz = wsum32(pz) * ik;
  float cx = px - mx, cy = py - my, cz = pz - mz;
  float a00 = wsum32(act ? cx * cx : 0.f) * ik1;
  float a01 = wsum32(act ? cx * cy : 0.f) * ik1;
  float a02 = wsum32(act ? cx * cz : 0.f) * ik1;
  float a11 = wsum32(act ? cy * cy : 0.f) * ik1;
  float a12 = wsum32(act ? cy * cz : 0.f) * ik1;
  float a22 = wsum32(act ? cz * cz : 0.f) * ik1;
  float R[9];
  eig3_sym(a00, a01, a02, a11, a12, a22, R);
  float qx = cx * R[0] + cy * R[3] + cz * R[6];
  float qy = cx * R[1] + cy * R[4] + cz * R[7];
  float qz = cx * R[2] + cy * R[5] + cz * R[8];
  float skx = wsum32(act ? qx * qx * qx : 0.f);
  float sky = wsum32(act ? qy * qy * qy : 0.f);
  float skz = wsum32(act ? qz * qz * qz : 0.f);
  float s0 = (skx >= 0.f) ? 1.f : -1.f;
  float s1 = (sky >= 0.f) ? 1.f : -1.f;
  float s2 = (skz >= 0.f) ? 1.f : -1.f;
  int neg = (s0 < 0.f) + (s1 < 0.f) + (s2 < 0.f);
  if (neg & 1) s2 = -s2;
  qx *= s0; qy *= s1; qz *= s2;
  float key = act ? qx : 1e30f;
  int payload = lane;
  for (int size = 2; size <= 32; size <<= 1) {
    for (int stride = size >> 1; stride > 0; stride >>= 1) {
      float okey = __shfl_xor(key, stride, 32);
      int   opay = __shfl_xor(payload, stride, 32);
      bool lower = (lane & stride) == 0;
      bool up    = (lane & size) == 0;
      bool takeMin = (lower == up);
      bool sw = takeMin ? (okey < key) : (okey > key);
      if (sw) { key = okey; payload = opay; }
    }
  }
  float ox = __shfl(qx, payload, 32);
  float oy = __shfl(qy, payload, 32);
  float oz = __shfl(qz, payload, 32);
  int   on = __shfl(nbr, payload, 32);
  int stride = 3 + C;
  if (act) {
    bf16_t* row = P + patch * cip + (size_t)lane * stride;
    row[0] = (bf16_t)ox; row[1] = (bf16_t)oy; row[2] = (bf16_t)oz;
    size_t gg = (size_t)b * N_PTS + on;
    for (int c = 0; c < C; ++c)
      row[3 + c] = feat[(size_t)c * BN_TOT + gg];
  }
  int ci = K_NN * stride;
  for (int i = ci + lane; i < cip; i += 32) P[patch * cip + i] = (bf16_t)0.f;
}

// ---------------- 5) weight f32 -> padded bf16 ----------------
__global__ void wconv_kernel(const float* __restrict__ W, int co, int ci, int cip,
                             bf16_t* __restrict__ Wb) {
  int i = blockIdx.x * blockDim.x + threadIdx.x;
  if (i >= co * cip) return;
  int r = i / cip, c = i % cip;
  Wb[i] = (bf16_t)((c < ci) ? W[(size_t)r * ci + c] : 0.f);
}

// ---------------- 6) WMMA GEMM, LDS-staged B strip (async copy), fused BN+leaky ----------
template <int MT>
__global__ void gemm_bn_lds_kernel(const bf16_t* __restrict__ Wb, const bf16_t* __restrict__ P,
                                   const float* __restrict__ scale, const float* __restrict__ bias,
                                   bf16_t* __restrict__ Xout, int cip) {
  constexpr int KC = 64;                              // K elements per chunk
  __shared__ __align__(32) bf16_t btile[2][16 * KC];  // 2 x 2 KB
  int lane = threadIdx.x & 31;
  int wave = threadIdx.x >> 5;                        // 0..3
  int n0 = blockIdx.x * 16;
  int mwave0 = (blockIdx.y * 4 + wave) * (MT * 16);
  v8f acc[MT];
#pragma unroll
  for (int i = 0; i < MT; ++i) acc[i] = (v8f){};

  const int t = threadIdx.x;                          // 128 threads x 16B = one 2KB chunk
  const int scol = t >> 3;                            // 8 threads per column
  const int ssub = (t & 7) * 16;                      // byte offset within the column's K run
  const char* gstage = (const char*)(P + (size_t)(n0 + scol) * cip) + ssub;
  char* lstage0 = (char*)&btile[0][scol * KC] + ssub;
  char* lstage1 = (char*)&btile[1][scol * KC] + ssub;

  const int koff = (lane >> 4) * 16;
  const bf16_t* aRow = Wb + (size_t)(lane & 15) * cip + koff;
  const bf16_t* bRow0 = &btile[0][(lane & 15) * KC + koff];
  const bf16_t* bRow1 = &btile[1][(lane & 15) * KC + koff];

  int nchunks = cip / KC;
  async_copy16(gstage, lstage0);                      // stage chunk 0
  for (int c = 0; c < nchunks; ++c) {
    int buf = c & 1;
    bool more = (c + 1) < nchunks;
    if (more) {                                       // stage chunk c+1 into other buffer
      async_copy16(gstage + (size_t)(c + 1) * KC * 2, buf ? lstage0 : lstage1);
      async_wait_le(1);
    } else {
      async_wait_le(0);
    }
    __syncthreads();                                  // chunk c visible to all waves
    const bf16_t* bR = buf ? bRow1 : bRow0;
    v16bf b0 = *(const v16bf*)(bR);
    v16bf b1 = *(const v16bf*)(bR + 32);
    size_t kc0 = (size_t)c * KC;
    __builtin_prefetch(aRow + (size_t)mwave0 * cip + kc0 + 256, 0, 1);
#pragma unroll
    for (int mt = 0; mt < MT; ++mt) {
      const bf16_t* aptr = aRow + (size_t)(mwave0 + mt * 16) * cip + kc0;
      v16bf a0 = *(const v16bf*)(aptr);
      v16bf a1 = *(const v16bf*)(aptr + 32);
      acc[mt] = __builtin_amdgcn_wmma_f32_16x16x32_bf16(false, a0, false, b0,
                                                        (short)0, acc[mt], false, false);
      acc[mt] = __builtin_amdgcn_wmma_f32_16x16x32_bf16(false, a1, false, b1,
                                                        (short)0, acc[mt], false, false);
    }
    __syncthreads();                                  // done reading buf before re-stage
  }

  int col = n0 + (lane & 15);
#pragma unroll
  for (int mt = 0; mt < MT; ++mt) {
#pragma unroll
    for (int v = 0; v < 8; ++v) {
      int m = mwave0 + mt * 16 + v + 8 * (lane >> 4);
      float y = acc[mt][v] * scale[m] + bias[m];
      y = (y > 0.f) ? y : LRELU * y;
      Xout[(size_t)m * BN_TOT + col] = (bf16_t)y;
    }
  }
}

// ---------------- 7) concat x1..x4 into point-major [bn][512] ----------------
__global__ void concat_kernel(const bf16_t* __restrict__ X1, const bf16_t* __restrict__ X2,
                              const bf16_t* __restrict__ X3, const bf16_t* __restrict__ X4,
                              bf16_t* __restrict__ XC) {
  long long i = (long long)blockIdx.x * blockDim.x + threadIdx.x;
  if (i >= (long long)BN_TOT * 512) return;
  int bn = (int)(i >> 9), c = (int)(i & 511);
  bf16_t v;
  if (c < 64)        v = X1[(size_t)c * BN_TOT + bn];
  else if (c < 128)  v = X2[(size_t)(c - 64) * BN_TOT + bn];
  else if (c < 256)  v = X3[(size_t)(c - 128) * BN_TOT + bn];
  else               v = X4[(size_t)(c - 256) * BN_TOT + bn];
  XC[(size_t)bn * 512 + c] = v;
}

// ---------------- 8) global max+mean pool ----------------
__global__ void pool_kernel(const bf16_t* __restrict__ X5, float* __restrict__ xp) {
  const int CO = 1024;
  int lane = threadIdx.x & 31;
  int wid = (blockIdx.x * blockDim.x + threadIdx.x) >> 5;
  if (wid >= B_SZ * CO) return;
  int bb = wid / CO, c = wid % CO;
  const bf16_t* row = X5 + (size_t)c * BN_TOT + (size_t)bb * N_PTS;
  float mx = -1e30f, sm = 0.f;
  for (int j = lane; j < N_PTS; j += 32) {
    float v = (float)row[j];
    mx = fmaxf(mx, v); sm += v;
  }
  for (int m = 16; m; m >>= 1) {
    mx = fmaxf(mx, __shfl_xor(mx, m, 32));
    sm += __shfl_xor(sm, m, 32);
  }
  if (lane == 0) {
    xp[(size_t)bb * 2 * CO + c]      = mx;
    xp[(size_t)bb * 2 * CO + CO + c] = sm / N_PTS;
  }
}

// ---------------- 9) small FC (one wave / output element) ----------------
__global__ void fc_kernel(const float* __restrict__ in, const float* __restrict__ W,
                          const float* __restrict__ badd, const float* __restrict__ s,
                          const float* __restrict__ b2, float* __restrict__ out,
                          int O, int L, int leaky) {
  int lane = threadIdx.x & 31;
  int wid = (blockIdx.x * blockDim.x + threadIdx.x) >> 5;
  if (wid >= B_SZ * O) return;
  int bb = wid / O, o = wid % O;
  const float* iv = in + (size_t)bb * L;
  const float* wv = W + (size_t)o * L;
  float acc = 0.f;
  for (int j = lane; j < L; j += 32) acc += iv[j] * wv[j];
  for (int m = 16; m; m >>= 1) acc += __shfl_xor(acc, m, 32);
  if (lane == 0) {
    float y = acc;
    if (badd) y += badd[o];
    if (s)    y = y * s[o] + b2[o];
    if (leaky) y = y > 0.f ? y : LRELU * y;
    out[(size_t)bb * O + o] = y;
  }
}

// ---------------- launch ----------------
extern "C" void kernel_launch(void* const* d_in, const int* in_sizes, int n_in,
                              void* d_out, int out_size, void* d_ws, size_t ws_size,
                              hipStream_t stream) {
  (void)in_sizes; (void)n_in; (void)out_size; (void)ws_size;
  const float* x  = (const float*)d_in[0];
  const float* W1 = (const float*)d_in[1],  *s1 = (const float*)d_in[2],  *b1 = (const float*)d_in[3];
  const float* W2 = (const float*)d_in[4],  *s2 = (const float*)d_in[5],  *b2 = (const float*)d_in[6];
  const float* W3 = (const float*)d_in[7],  *s3 = (const float*)d_in[8],  *b3 = (const float*)d_in[9];
  const float* W4 = (const float*)d_in[10], *s4 = (const float*)d_in[11], *b4 = (const float*)d_in[12];
  const float* W5 = (const float*)d_in[13], *s5 = (const float*)d_in[14], *b5 = (const float*)d_in[15];
  const float* Wl1 = (const float*)d_in[16], *s6 = (const float*)d_in[17], *b6 = (const float*)d_in[18];
  const float* Wl2 = (const float*)d_in[19], *bl2 = (const float*)d_in[20];
  const float* s7  = (const float*)d_in[21], *b7  = (const float*)d_in[22];
  const float* Wl3 = (const float*)d_in[23], *bl3 = (const float*)d_in[24];
  float* out = (float*)d_out;

  char* wsb = (char*)d_ws;
  size_t off = 0;
  auto alloc = [&](size_t bytes) -> void* {
    void* p = wsb + off;
    off = (off + bytes + 255) & ~(size_t)255;
    return p;
  };
  float*  canon = (float*)alloc((size_t)BN_TOT * 3 * 4);
  bf16_t* XT    = (bf16_t*)alloc((size_t)BN_TOT * 128 * 2);
  float*  norms = (float*)alloc((size_t)BN_TOT * 4);
  int*    idx   = (int*)alloc((size_t)BN_TOT * K_NN * 4);
  bf16_t* X1 = (bf16_t*)alloc((size_t)64  * BN_TOT * 2);
  bf16_t* X2 = (bf16_t*)alloc((size_t)64  * BN_TOT * 2);
  bf16_t* X3 = (bf16_t*)alloc((size_t)128 * BN_TOT * 2);
  bf16_t* X4 = (bf16_t*)alloc((size_t)256 * BN_TOT * 2);
  bf16_t* P  = (bf16_t*)alloc((size_t)BN_TOT * 2624 * 2);   // reused for XC
  bf16_t* X5 = (bf16_t*)alloc((size_t)1024 * BN_TOT * 2);
  bf16_t* Wb = (bf16_t*)alloc((size_t)1024 * 2624 * 2);     // reused per layer
  float*  xp = (float*)alloc((size_t)B_SZ * 2048 * 4);
  float*  h1 = (float*)alloc((size_t)B_SZ * 512 * 4);
  float*  h2 = (float*)alloc((size_t)B_SZ * 256 * 4);

  dim3 knnGrid(N_PTS / 16, B_SZ);
  int patchBlocks = BN_TOT / 4;
  int nTiles = BN_TOT / 16;

  // Stage 0: cloud PCA canonicalization
  cloud_pca_kernel<<<B_SZ, 256, 0, stream>>>(x, canon);

  // Layer 1 (features = canon pts, C=3 -> cp=32; ci=60 -> cip=64, co=64, MT=1)
  make_xt_pts_kernel<<<BN_TOT / 256, 256, 0, stream>>>(canon, XT, norms);
  knn_kernel<<<knnGrid, 128, 0, stream>>>(XT, norms, 32, idx);
  patch_kernel<<<patchBlocks, 128, 0, stream>>>(canon, idx, nullptr, 0, P, 64);
  wconv_kernel<<<(64 * 64 + 255) / 256, 256, 0, stream>>>(W1, 64, 60, 64, Wb);
  gemm_bn_lds_kernel<1><<<dim3(nTiles, 1), 128, 0, stream>>>(Wb, P, s1, b1, X1, 64);

  // Layer 2 (C=64, ci=1340 -> cip=1344, co=64, MT=1)
  make_xt_kernel<<<BN_TOT / 256, 256, 0, stream>>>(X1, 64, 64, XT, norms);
  knn_kernel<<<knnGrid, 128, 0, stream>>>(XT, norms, 64, idx);
  patch_kernel<<<patchBlocks, 128, 0, stream>>>(canon, idx, X1, 64, P, 1344);
  wconv_kernel<<<(64 * 1344 + 255) / 256, 256, 0, stream>>>(W2, 64, 1340, 1344, Wb);
  gemm_bn_lds_kernel<1><<<dim3(nTiles, 1), 128, 0, stream>>>(Wb, P, s2, b2, X2, 1344);

  // Layer 3 (C=64, ci=1340 -> cip=1344, co=128, MT=2)
  make_xt_kernel<<<BN_TOT / 256, 256, 0, stream>>>(X2, 64, 64, XT, norms);
  knn_kernel<<<knnGrid, 128, 0, stream>>>(XT, norms, 64, idx);
  patch_kernel<<<patchBlocks, 128, 0, stream>>>(canon, idx, X2, 64, P, 1344);
  wconv_kernel<<<(128 * 1344 + 255) / 256, 256, 0, stream>>>(W3, 128, 1340, 1344, Wb);
  gemm_bn_lds_kernel<2><<<dim3(nTiles, 1), 128, 0, stream>>>(Wb, P, s3, b3, X3, 1344);

  // Layer 4 (C=128, ci=2620 -> cip=2624, co=256, MT=4)
  make_xt_kernel<<<BN_TOT / 256, 256, 0, stream>>>(X3, 128, 128, XT, norms);
  knn_kernel<<<knnGrid, 128, 0, stream>>>(XT, norms, 128, idx);
  patch_kernel<<<patchBlocks, 128, 0, stream>>>(canon, idx, X3, 128, P, 2624);
  wconv_kernel<<<(256 * 2624 + 255) / 256, 256, 0, stream>>>(W4, 256, 2620, 2624, Wb);
  gemm_bn_lds_kernel<4><<<dim3(nTiles, 1), 128, 0, stream>>>(Wb, P, s4, b4, X4, 2624);

  // Layer 5: concat (512) -> 1024; MT=4 x grid.y=4 covers co=1024
  long long ce = (long long)BN_TOT * 512;
  concat_kernel<<<(unsigned)((ce + 255) / 256), 256, 0, stream>>>(X1, X2, X3, X4, P);
  wconv_kernel<<<(1024 * 512 + 255) / 256, 256, 0, stream>>>(W5, 1024, 512, 512, Wb);
  gemm_bn_lds_kernel<4><<<dim3(nTiles, 4), 128, 0, stream>>>(Wb, P, s5, b5, X5, 512);

  // Pool + classifier head
  pool_kernel<<<(B_SZ * 1024) / 4, 128, 0, stream>>>(X5, xp);
  fc_kernel<<<(B_SZ * 512 * 32 + 127) / 128, 128, 0, stream>>>(xp, Wl1, nullptr, s6, b6, h1, 512, 2048, 1);
  fc_kernel<<<(B_SZ * 256 * 32 + 127) / 128, 128, 0, stream>>>(h1, Wl2, bl2, s7, b7, h2, 256, 512, 1);
  fc_kernel<<<(B_SZ * 40 * 32 + 127) / 128, 128, 0, stream>>>(h2, Wl3, bl3, nullptr, nullptr, out, 40, 256, 0);
}
